// IQALoss_79577154060421
// MI455X (gfx1250) — matrix operand order repaired and checked by reading, not compiled, not run on hardware
//
#include <hip/hip_runtime.h>
#include <math.h>

typedef __attribute__((ext_vector_type(16))) _Float16 v16h;
typedef __attribute__((ext_vector_type(8)))  _Float16 v8h;
typedef __attribute__((ext_vector_type(8)))  float    v8f;

#define B_N 4096
#define D_N 512
#define TEMP_INV 10.0f
#define EPS_F 1e-8f
#define MARGIN_F 0.1f

// ---------------- utility: block reductions ----------------
__device__ __forceinline__ float warp_sum(float v) {
    #pragma unroll
    for (int m = 16; m > 0; m >>= 1) v += __shfl_xor(v, m, 32);
    return v;
}

// ---------------- kernel 1: zero accumulators ----------------
__global__ void zero_kernel(float* p, int n) {
    int i = blockIdx.x * blockDim.x + threadIdx.x;
    if (i < n) p[i] = 0.0f;
}

// ---------------- kernel 2: Huber + PLCC moment sums ----------------
// scal[0]=huber_q_sum scal[1]=huber_id_sum
// scal[2..6]  = q:  sx, sy, sxx, syy, sxy
// scal[7..11] = id: sx, sy, sxx, syy, sxy
__global__ void moments_kernel(const float* __restrict__ qp,
                               const float* __restrict__ ip,
                               const float* __restrict__ qg,
                               const float* __restrict__ ig,
                               float* __restrict__ scal) {
    float s[12];
    #pragma unroll
    for (int k = 0; k < 12; ++k) s[k] = 0.0f;

    for (int i = threadIdx.x; i < B_N; i += blockDim.x) {
        float a = qp[i], b = qg[i];
        float d = fabsf(a - b);
        s[0] += (d < 1.0f) ? 0.5f * d * d : d - 0.5f;
        s[2] += a; s[3] += b; s[4] += a * a; s[5] += b * b; s[6] += a * b;
        a = ip[i]; b = ig[i];
        d = fabsf(a - b);
        s[1] += (d < 1.0f) ? 0.5f * d * d : d - 0.5f;
        s[7] += a; s[8] += b; s[9] += a * a; s[10] += b * b; s[11] += a * b;
    }

    __shared__ float sm[8];
    const int lane = threadIdx.x & 31, w = threadIdx.x >> 5;
    for (int k = 0; k < 12; ++k) {
        __syncthreads();
        float v = warp_sum(s[k]);
        if (lane == 0) sm[w] = v;
        __syncthreads();
        if (threadIdx.x == 0) {
            float t = 0.0f;
            #pragma unroll
            for (int q = 0; q < 8; ++q) t += sm[q];
            scal[k] = t;
        }
    }
}

// ---------------- kernel 3: row-normalize feats -> f16 ----------------
__global__ __launch_bounds__(256) void normalize_kernel(const float* __restrict__ feats,
                                                        _Float16* __restrict__ fn) {
    const int row = blockIdx.x;
    const float* src = feats + (size_t)row * D_N;
    float v0 = src[threadIdx.x];
    float v1 = src[threadIdx.x + 256];
    float ss = v0 * v0 + v1 * v1;

    __shared__ float sm[8];
    __shared__ float total;
    const int lane = threadIdx.x & 31, w = threadIdx.x >> 5;
    ss = warp_sum(ss);
    if (lane == 0) sm[w] = ss;
    __syncthreads();
    if (threadIdx.x == 0) {
        float t = 0.0f;
        #pragma unroll
        for (int q = 0; q < 8; ++q) t += sm[q];
        total = t;
    }
    __syncthreads();
    const float inv = rsqrtf(total);
    fn[(size_t)row * D_N + threadIdx.x]       = (_Float16)(v0 * inv);
    fn[(size_t)row * D_N + threadIdx.x + 256] = (_Float16)(v1 * inv);
}

// ---------------- kernel 4: ranking loss + same-id counts ----------------
// Cnt[i] = (# j with raw_ids[j]==raw_ids[i]) - 1   (positives excluding self)
// scal[12] += rank loss sum, scal[13] += rank pair count
__global__ __launch_bounds__(256) void rank_count_kernel(const float* __restrict__ id_pred,
                                                         const float* __restrict__ id_gt,
                                                         const int* __restrict__ rid,
                                                         float* __restrict__ Cnt,
                                                         float* __restrict__ scal) {
    const int i = blockIdx.x;
    const int   ri = rid[i];
    const float gi = id_gt[i];
    const float pi = id_pred[i];

    float cnt_same = 0.0f, ls = 0.0f, mc = 0.0f;
    for (int j = threadIdx.x; j < B_N; j += 256) {
        const bool same = (rid[j] == ri);
        cnt_same += same ? 1.0f : 0.0f;
        if (same && (gi > id_gt[j] + 1e-6f)) {
            float r = MARGIN_F - (pi - id_pred[j]);
            ls += (r > 0.0f) ? r : 0.0f;
            mc += 1.0f;
        }
    }
    __shared__ float sm[8][3];
    const int lane = threadIdx.x & 31, w = threadIdx.x >> 5;
    cnt_same = warp_sum(cnt_same);
    ls = warp_sum(ls);
    mc = warp_sum(mc);
    if (lane == 0) { sm[w][0] = cnt_same; sm[w][1] = ls; sm[w][2] = mc; }
    __syncthreads();
    if (threadIdx.x == 0) {
        float c = 0, l = 0, m = 0;
        #pragma unroll
        for (int q = 0; q < 8; ++q) { c += sm[q][0]; l += sm[q][1]; m += sm[q][2]; }
        Cnt[i] = c - 1.0f;
        atomicAdd(&scal[12], l);
        atomicAdd(&scal[13], m);
    }
}

// ---------------- kernel 5: WMMA GEMM + fused supcon row reductions ----------------
// Per 128x128 tile of logits = (fn fn^T) * 10:
//   S1[i] += sum_j exp(logit_ij)   (incl. diagonal; removed in finalize)
//   S2[i] += sum_{j: same id} logit_ij (incl. diagonal; removed in finalize)
__global__ __launch_bounds__(256) void supcon_gemm_kernel(const _Float16* __restrict__ fn,
                                                          const int* __restrict__ rid,
                                                          float* __restrict__ S1,
                                                          float* __restrict__ S2) {
    __shared__ _Float16 At[128][40];   // rows i0..i0+127, 32 K-values (+pad)
    __shared__ _Float16 Bt[128][40];   // rows j0..j0+127, 32 K-values (+pad)
    __shared__ float S1l[128], S2l[128];
    __shared__ int idsI[128], idsJ[128];

    const int i0 = blockIdx.y * 128;
    const int j0 = blockIdx.x * 128;
    const int t  = threadIdx.x;

    if (t < 128) { S1l[t] = 0.0f; S2l[t] = 0.0f; idsI[t] = rid[i0 + t]; }
    else         { int u = t - 128; idsJ[u] = rid[j0 + u]; }

    const int lane  = t & 31;
    const int w     = t >> 5;             // 8 waves of 32 (wave32)
    const int mbase = (w >> 2) * 64;      // 2 wave-rows
    const int nbase = (w & 3) * 32;       // 4 wave-cols

    v8f acc[4][2];
    #pragma unroll
    for (int mt = 0; mt < 4; ++mt)
        #pragma unroll
        for (int nt = 0; nt < 2; ++nt) { v8f z = {}; acc[mt][nt] = z; }

    const int lr = t >> 1;            // LDS-fill row 0..127
    const int lc = (t & 1) * 16;      // LDS-fill col chunk (f16 units)

    union Frag { v16h v; v8h h[2]; };

    for (int k0 = 0; k0 < D_N; k0 += 32) {
        __syncthreads();
        // Stage A and B tiles (each thread: 2x16B contiguous per tile)
        {
            const uint4* sa = reinterpret_cast<const uint4*>(fn + (size_t)(i0 + lr) * D_N + k0 + lc);
            uint4* da = reinterpret_cast<uint4*>(&At[lr][lc]);
            da[0] = sa[0]; da[1] = sa[1];
            const uint4* sb = reinterpret_cast<const uint4*>(fn + (size_t)(j0 + lr) * D_N + k0 + lc);
            uint4* db = reinterpret_cast<uint4*>(&Bt[lr][lc]);
            db[0] = sb[0]; db[1] = sb[1];
        }
        __syncthreads();

        // B fragments: 32x16 f16, lane = N (=col j), halves hold K chunks {0..7,16..23}/{8..15,24..31}
        Frag bf[2];
        const int nrow = nbase + (lane & 15);
        const int kh   = (lane >> 4) * 8;
        bf[0].h[0] = *(const v8h*)&Bt[nrow][kh];
        bf[0].h[1] = *(const v8h*)&Bt[nrow][kh + 16];
        bf[1].h[0] = *(const v8h*)&Bt[nrow + 16][kh];
        bf[1].h[1] = *(const v8h*)&Bt[nrow + 16][kh + 16];

        #pragma unroll
        for (int mt = 0; mt < 4; ++mt) {
            // A fragment: 16x32 f16, lane%16 = M, halves hold K chunks {0..7,16..23}/{8..15,24..31}
            Frag af;
            const int mr = mbase + mt * 16 + (lane & 15);
            const int kc = (lane >> 4) * 8;
            af.h[0] = *(const v8h*)&At[mr][kc];
            af.h[1] = *(const v8h*)&At[mr][16 + kc];
            acc[mt][0] = __builtin_amdgcn_wmma_f32_16x16x32_f16(
                false, af.v, false, bf[0].v, (short)0, acc[mt][0], false, false);
            acc[mt][1] = __builtin_amdgcn_wmma_f32_16x16x32_f16(
                false, af.v, false, bf[1].v, (short)0, acc[mt][1], false, false);
        }
    }
    __syncthreads();

    // Fused epilogue: C layout -> VGPR g: M = g + 8*(lane>=16), N = lane&15
    #pragma unroll
    for (int mt = 0; mt < 4; ++mt) {
        #pragma unroll
        for (int nt = 0; nt < 2; ++nt) {
            v8f c = acc[mt][nt];
            #pragma unroll
            for (int g = 0; g < 8; ++g) {
                const int M = mbase + mt * 16 + g + ((lane >> 4) << 3);
                const int N = nbase + nt * 16 + (lane & 15);
                const float logit = c[g] * TEMP_INV;
                float e  = __expf(logit);
                float s2 = (idsI[M] == idsJ[N]) ? logit : 0.0f;
                // reduce across the 16 lanes (columns) sharing this row
                #pragma unroll
                for (int m = 8; m > 0; m >>= 1) {
                    e  += __shfl_xor(e,  m, 32);
                    s2 += __shfl_xor(s2, m, 32);
                }
                if ((lane & 15) == 0) {
                    atomicAdd(&S1l[M], e);
                    atomicAdd(&S2l[M], s2);
                }
            }
        }
    }
    __syncthreads();
    if (t < 128) {
        atomicAdd(&S1[i0 + t], S1l[t]);
        atomicAdd(&S2[i0 + t], S2l[t]);
    }
}

// ---------------- kernel 6: finalize scalar loss ----------------
__global__ void finalize_kernel(const float* __restrict__ S1,
                                const float* __restrict__ S2,
                                const float* __restrict__ Cnt,
                                const float* __restrict__ scal,
                                float* __restrict__ out) {
    // supcon per-row: mlpp_i = (sum_pos logits - C*log(denom+eps)) / (C+eps)
    const float diag_e = __expf(TEMP_INV);   // exp(l_ii), l_ii = 1/TEMP after normalization
    float sup = 0.0f;
    for (int i = threadIdx.x; i < B_N; i += 256) {
        const float denom = S1[i] - diag_e;                // exclude eye from exp sum
        const float lp = __logf(denom + EPS_F);
        const float s2 = S2[i] - TEMP_INV;                 // exclude eye from positive sum
        const float Cf = Cnt[i];
        sup += (s2 - Cf * lp) / (Cf + EPS_F);
    }
    __shared__ float sm[8];
    const int lane = threadIdx.x & 31, w = threadIdx.x >> 5;
    sup = warp_sum(sup);
    if (lane == 0) sm[w] = sup;
    __syncthreads();
    if (threadIdx.x == 0) {
        float sup_total = 0.0f;
        #pragma unroll
        for (int q = 0; q < 8; ++q) sup_total += sm[q];

        const float invB = 1.0f / (float)B_N;
        const float hq  = scal[0] * invB;
        const float hid = scal[1] * invB;

        float mx = scal[2] * invB, my = scal[3] * invB;
        float vx = scal[4] * invB - mx * mx;
        float vy = scal[5] * invB - my * my;
        float cv = scal[6] * invB - mx * my;
        const float corr_q = cv / (sqrtf(vx) * sqrtf(vy) + EPS_F);

        mx = scal[7] * invB; my = scal[8] * invB;
        vx = scal[9] * invB - mx * mx;
        vy = scal[10] * invB - my * my;
        cv = scal[11] * invB - mx * my;
        const float corr_id = cv / (sqrtf(vx) * sqrtf(vy) + EPS_F);

        const float plcc = (1.0f - corr_q) + (1.0f - corr_id);
        const float loss_sup = -(sup_total * invB);
        const float rs = scal[12], rc = scal[13];
        const float rank = (rc > 0.0f) ? rs / fmaxf(rc, 1.0f) : 0.0f;

        out[0] = hq + hid + 0.1f * plcc + 0.05f * loss_sup + 0.05f * rank;
    }
}

// ---------------- host-side launch ----------------
extern "C" void kernel_launch(void* const* d_in, const int* in_sizes, int n_in,
                              void* d_out, int out_size, void* d_ws, size_t ws_size,
                              hipStream_t stream) {
    const float* q_pred  = (const float*)d_in[0];
    const float* id_pred = (const float*)d_in[1];
    const float* q_gt    = (const float*)d_in[2];
    const float* id_gt   = (const float*)d_in[3];
    const float* feats   = (const float*)d_in[4];
    const int*   raw_ids = (const int*)d_in[5];
    float* out = (float*)d_out;

    char* ws = (char*)d_ws;
    _Float16* fn = (_Float16*)ws;                                  // 4 MB
    float* S1   = (float*)(ws + (size_t)B_N * D_N * sizeof(_Float16));
    float* S2   = S1 + B_N;
    float* Cnt  = S2 + B_N;
    float* scal = Cnt + B_N;                                       // 16 floats

    // Zero S1/S2/Cnt/scal every call (graph replays don't re-init ws)
    const int nzero = 3 * B_N + 16;
    zero_kernel<<<(nzero + 255) / 256, 256, 0, stream>>>(S1, nzero);

    moments_kernel<<<1, 256, 0, stream>>>(q_pred, id_pred, q_gt, id_gt, scal);
    normalize_kernel<<<B_N, 256, 0, stream>>>(feats, fn);
    rank_count_kernel<<<B_N, 256, 0, stream>>>(id_pred, id_gt, raw_ids, Cnt, scal);

    dim3 gemm_grid(B_N / 128, B_N / 128);   // 32 x 32
    supcon_gemm_kernel<<<gemm_grid, 256, 0, stream>>>(fn, raw_ids, S1, S2);

    finalize_kernel<<<1, 256, 0, stream>>>(S1, S2, Cnt, scal, out);
}
